// Model_23201413333024
// MI455X (gfx1250) — matrix operand order repaired and checked by reading, not compiled, not run on hardware
//
#include <hip/hip_runtime.h>
#include <hip/hip_bf16.h>

#define N_OPS   100000
#define N_DEV   256
#define E_PREV  400000
#define E_SUCC  400000
#define E_LINK  65536
#define E_PLACE 1000000
#define E_SERVE 1000000

// flattened param offsets (f32 elements), jax tree_flatten order (sorted dict keys, b<w)
#define P_DEV_B      0
#define P_DEV_W      40
#define P_ET_LINK_B  680
#define P_ET_LINK_W  688
#define P_ET_PLACE_B 784
#define P_ET_PLACE_W 792
#define P_ET_PREV_B  824
#define P_ET_PREV_W  832
#define P_ET_SERVE_B 896
#define P_ET_SERVE_W 904
#define P_ET_SUCC_B  936
#define P_ET_SUCC_W  944
#define P_FINAL_B    1008
#define P_FINAL_W    1009
#define P_GCONV      1097   // + l*9800 + pos*1960 ; bias at +0, W[48][40] at +40
#define P_OPEMB      40297
#define P_OPT_B      42345
#define P_OPT_W      42385

typedef __attribute__((ext_vector_type(16))) __bf16 v16bf;
typedef __attribute__((ext_vector_type(8)))  float  v8f;

union AU { v16bf v; unsigned short h[16]; uint4 q[2]; };

__device__ __forceinline__ unsigned short f2bf(float x) {
  union { float f; unsigned int u; } c; c.f = x;
  unsigned int r = c.u + 0x7FFFu + ((c.u >> 16) & 1u);   // round-to-nearest-even
  return (unsigned short)(r >> 16);
}
__device__ __forceinline__ float eluf(float x) { return x > 0.f ? x : expm1f(x); }

// ---------------------------------------------------------------- counts
__global__ void count_kernel(const int* __restrict__ dst, unsigned int* __restrict__ cnt, int E) {
  int i = blockIdx.x * 256 + threadIdx.x;
  if (i < E) atomicAdd(&cnt[dst[i]], 1u);
}
__global__ void invert_kernel(float* __restrict__ buf, int N) {
  int i = blockIdx.x * 256 + threadIdx.x;
  if (i < N) {
    unsigned int c = ((unsigned int*)buf)[i];
    buf[i] = 1.0f / (float)(c > 0u ? c : 1u);
  }
}

// ---------------------------------------------------------------- pack gconv W into WMMA B-fragments
// frag f = ((l*5+et)*3+nt)*2+kt ; layout per fragment: [32 lanes][16 bf16]
// B 32x16 bf16: lane -> N (lane&15), lanes>=16 hold K+8; elems 0..7 -> K 0..7, elems 8..15 -> K 16..23 (even K low half)
__global__ void pack_weights_kernel(const float* __restrict__ params, unsigned short* __restrict__ wpack) {
  int f = blockIdx.x;            // 0..119
  int lane = threadIdx.x;        // 0..31
  int kt = f & 1; int t = f >> 1;
  int nt = t % 3; int u = t / 3;
  int et = u % 5; int l = u / 5;
  int pos = (et == 0) ? 0 : (et == 1) ? 2 : (et == 2) ? 4 : (et == 3) ? 1 : 3; // alpha sort pos
  const float* W = params + P_GCONV + l * 9800 + pos * 1960 + 40;  // [48][40] row-major (k,n)
  int n  = nt * 16 + (lane & 15);
  int kb = kt * 32 + (lane >> 4) * 8;
  unsigned short* out = wpack + ((long)f * 32 + lane) * 16;
  for (int j = 0; j < 16; ++j) {
    int k = kb + (j < 8 ? j : 8 + j);
    float v = (k < 48 && n < 40) ? W[k * 40 + n] : 0.f;
    out[j] = f2bf(v);
  }
}

// ---------------------------------------------------------------- edge feature transform (din->8, ELU, bf16 out)
__global__ void edge_trans_kernel(const float* __restrict__ x, const float* __restrict__ params,
                                  int w_off, int b_off, int din,
                                  unsigned short* __restrict__ out, int E) {
  __shared__ float ws[12 * 8];
  __shared__ float bs[8];
  if (threadIdx.x < din * 8) ws[threadIdx.x] = params[w_off + threadIdx.x];
  if (threadIdx.x < 8)       bs[threadIdx.x] = params[b_off + threadIdx.x];
  __syncthreads();
  long e = (long)blockIdx.x * 256 + threadIdx.x;
  if (e >= E) return;
  float a[8];
#pragma unroll
  for (int j = 0; j < 8; ++j) a[j] = bs[j];
  for (int k = 0; k < din; ++k) {
    float xv = x[e * din + k];
#pragma unroll
    for (int j = 0; j < 8; ++j) a[j] += xv * ws[k * 8 + j];
  }
  unsigned short h[8];
#pragma unroll
  for (int j = 0; j < 8; ++j) h[j] = f2bf(eluf(a[j]));
  *(uint4*)(out + e * 8) = *(uint4*)h;
}

// ---------------------------------------------------------------- node transforms (write f32 + bf16 mirror)
__global__ void op_trans_kernel(const float* __restrict__ opfeat, const float* __restrict__ ntp,
                                const int* __restrict__ types, const float* __restrict__ params,
                                float* __restrict__ out, unsigned short* __restrict__ out_bf, int N) {
  __shared__ float Ws[41 * 40 + 40];
  for (int i = threadIdx.x; i < 41 * 40; i += 256) Ws[i] = params[P_OPT_W + i];
  if (threadIdx.x < 40) Ws[41 * 40 + threadIdx.x] = params[P_OPT_B + threadIdx.x];
  __syncthreads();
  int op = blockIdx.x * 256 + threadIdx.x;
  if (op >= N) return;
  float x[41];
#pragma unroll
  for (int k = 0; k < 32; ++k) x[k] = opfeat[(long)op * 32 + k];
  const float* em = params + P_OPEMB + (long)types[op] * 8;
#pragma unroll
  for (int k = 0; k < 8; ++k) x[32 + k] = em[k];
  x[40] = ntp[op];
  float* o = out + (long)op * 40;
  unsigned short* ob = out_bf + (long)op * 40;
  for (int n = 0; n < 40; ++n) {
    float a = Ws[41 * 40 + n];
#pragma unroll
    for (int k = 0; k < 41; ++k) a += x[k] * Ws[k * 40 + n];
    float v = eluf(a);
    o[n] = v;
    ob[n] = f2bf(v);
  }
}

__global__ void dev_trans_kernel(const float* __restrict__ df, const float* __restrict__ params,
                                 float* __restrict__ out, unsigned short* __restrict__ out_bf) {
  __shared__ float Ws[16 * 40 + 40];
  for (int i = threadIdx.x; i < 16 * 40; i += 256) Ws[i] = params[P_DEV_W + i];
  if (threadIdx.x < 40) Ws[640 + threadIdx.x] = params[P_DEV_B + threadIdx.x];
  __syncthreads();
  int d = threadIdx.x;
  if (d >= N_DEV) return;
  float x[16];
#pragma unroll
  for (int k = 0; k < 16; ++k) x[k] = df[d * 16 + k];
  for (int n = 0; n < 40; ++n) {
    float a = Ws[640 + n];
#pragma unroll
    for (int k = 0; k < 16; ++k) a += x[k] * Ws[k * 40 + n];
    float v = eluf(a);
    out[d * 40 + n] = v;
    out_bf[d * 40 + n] = f2bf(v);
  }
}

// ---------------------------------------------------------------- gconv edge GEMM via bf16 WMMA
// Per wave: 16 edges. A = [gathered bf16 src feats(40) | bf16 edge feats(8) | pad] (16x64, 2 k-steps)
// B = pre-packed W fragments (64x48 padded), all 6 preloaded up front so the three WMMA pairs and
// atomic clauses issue back-to-back. C 16x16 f32, 3 n-tiles.
// Scatter-mean via scaled f32 atomics; pad edges clamp to edge E-1 with ic=0 (branchless +0.0 adds).
__global__ void __launch_bounds__(256)
gconv_gemm_kernel(const unsigned short* __restrict__ src_bf, const unsigned short* __restrict__ ef,
                  const int* __restrict__ s_idx, const int* __restrict__ d_idx,
                  const float* __restrict__ inv_cnt, const unsigned short* __restrict__ wpack,
                  const float* __restrict__ bias, float* __restrict__ acc, int E) {
  const int lane = threadIdx.x & 31;
  const int tile = blockIdx.x * 8 + (threadIdx.x >> 5);
  const long ebase = (long)tile * 16;
  if (ebase >= E) return;                    // uniform per wave: EXEC stays all-ones for WMMA
  const int row  = lane & 15;
  const int half = lane >> 4;
  const long e  = ebase + row;
  const long ec = e < E ? e : (long)(E - 1); // clamp (garbage rows never stored)
  const int s = s_idx[ec];
  const unsigned short* sf = src_bf + (long)s * 40;

  // A fragments: bf16 memory pairs are already (even K low half, odd K high half)
  AU a0, a1;
  a0.q[0] = *(const uint4*)(sf + half * 8);        // K = half*8 + 0..7
  a0.q[1] = *(const uint4*)(sf + 16 + half * 8);   // K = 16 + half*8 + 0..7
  const unsigned short* a1p = half ? (ef + ec * 8) // K = 40..47 (edge feats)
                                   : (sf + 32);    // K = 32..39 (node feats)
  a1.q[0] = *(const uint4*)a1p;
  a1.q[1] = make_uint4(0, 0, 0, 0);                // K = 48..63 zero pad

  // Preload all 6 B fragments (shared by the whole grid; L0-resident)
  AU b[3][2];
#pragma unroll
  for (int nt = 0; nt < 3; ++nt)
#pragma unroll
    for (int kt = 0; kt < 2; ++kt) {
      const unsigned short* w = wpack + ((nt * 2 + kt) * 32 + lane) * 16;
      b[nt][kt].q[0] = *(const uint4*)(w);
      b[nt][kt].q[1] = *(const uint4*)(w + 8);
    }

  // Bias values for the columns this lane owns (one per n-tile)
  float bn[3];
#pragma unroll
  for (int nt = 0; nt < 3; ++nt) {
    int n = nt * 16 + row;
    bn[nt] = bias[n < 40 ? n : 39];
  }

  int dloc[8]; float ic[8];                        // C rows this lane stores: M = half*8 + r
  if (ebase + 16 <= (long)E) {
    int4 d0 = *(const int4*)(d_idx + ebase + half * 8);
    int4 d1 = *(const int4*)(d_idx + ebase + half * 8 + 4);
    dloc[0] = d0.x; dloc[1] = d0.y; dloc[2] = d0.z; dloc[3] = d0.w;
    dloc[4] = d1.x; dloc[5] = d1.y; dloc[6] = d1.z; dloc[7] = d1.w;
#pragma unroll
    for (int r = 0; r < 8; ++r) ic[r] = inv_cnt[dloc[r]];
  } else {
#pragma unroll
    for (int r = 0; r < 8; ++r) {
      long er = ebase + half * 8 + r;
      long erc = er < E ? er : (long)(E - 1);
      dloc[r] = d_idx[erc];
      ic[r] = (er < E) ? inv_cnt[dloc[r]] : 0.f;   // pad edges contribute +0.0
    }
  }

#pragma unroll
  for (int nt = 0; nt < 3; ++nt) {
    v8f c = {};
    c = __builtin_amdgcn_wmma_f32_16x16x32_bf16(false, a0.v, false, b[nt][0].v, (short)0, c, false, false);
    c = __builtin_amdgcn_wmma_f32_16x16x32_bf16(false, a1.v, false, b[nt][1].v, (short)0, c, false, false);
    const int n = nt * 16 + row;                   // column held by this lane
    if (nt < 2 || row < 8) {                       // nt 0/1: no guard after unroll; nt 2: n<40
#pragma unroll
      for (int r = 0; r < 8; ++r)
        atomicAdd(acc + (long)dloc[r] * 40 + n, (c[r] + bn[nt]) * ic[r]);
    }
  }
}

// ---------------------------------------------------------------- residual + per-etype average + act
__global__ void combine_kernel(const float* __restrict__ cur, const float* __restrict__ acc,
                               float* __restrict__ nxt, unsigned short* __restrict__ nxt_bf,
                               float scale, int n, int do_elu) {
  int i = blockIdx.x * 256 + threadIdx.x;
  if (i < n) {
    float v = cur[i] + acc[i] * scale;
    v = do_elu ? eluf(v) : v;
    nxt[i] = v;
    nxt_bf[i] = f2bf(v);
  }
}

// ---------------------------------------------------------------- decoder: [opf|ef_place|devf] @ w(88) + b
__global__ void final_kernel(const float* __restrict__ opf, const float* __restrict__ devf,
                             const float* __restrict__ pfeat, const int* __restrict__ ps,
                             const int* __restrict__ pd, const float* __restrict__ params,
                             float* __restrict__ out, int E) {
  __shared__ float wf[89];
  __shared__ float wet[32];
  __shared__ float bet[8];
  if (threadIdx.x < 88) wf[threadIdx.x] = params[P_FINAL_W + threadIdx.x];
  if (threadIdx.x == 88) wf[88] = params[P_FINAL_B];
  if (threadIdx.x >= 96 && threadIdx.x < 128) wet[threadIdx.x - 96] = params[P_ET_PLACE_W + threadIdx.x - 96];
  if (threadIdx.x >= 128 && threadIdx.x < 136) bet[threadIdx.x - 128] = params[P_ET_PLACE_B + threadIdx.x - 128];
  __syncthreads();
  long e = (long)blockIdx.x * 256 + threadIdx.x;
  if (e >= E) return;
  const float* po = opf  + (long)ps[e] * 40;
  const float* pv = devf + (long)pd[e] * 40;
  float acc = wf[88];
#pragma unroll 8
  for (int k = 0; k < 40; ++k) acc += po[k] * wf[k];
  float x0 = pfeat[e * 4 + 0], x1 = pfeat[e * 4 + 1], x2 = pfeat[e * 4 + 2], x3 = pfeat[e * 4 + 3];
#pragma unroll
  for (int j = 0; j < 8; ++j) {
    float a = bet[j] + x0 * wet[j] + x1 * wet[8 + j] + x2 * wet[16 + j] + x3 * wet[24 + j];
    acc += eluf(a) * wf[40 + j];
  }
#pragma unroll 8
  for (int k = 0; k < 40; ++k) acc += pv[k] * wf[48 + k];
  out[e] = acc;
}

// ================================================================ host
extern "C" void kernel_launch(void* const* d_in, const int* in_sizes, int n_in,
                              void* d_out, int out_size, void* d_ws, size_t ws_size,
                              hipStream_t stream) {
  (void)in_sizes; (void)n_in; (void)out_size; (void)ws_size;
  const float* op_feats  = (const float*)d_in[0];
  const float* dev_feats = (const float*)d_in[1];
  const float* tensor_f  = (const float*)d_in[2];
  const float* link_f    = (const float*)d_in[3];
  const float* place_f   = (const float*)d_in[4];
  const float* ntp       = (const float*)d_in[5];
  const int*   op_types  = (const int*)d_in[6];
  const int* prev_s = (const int*)d_in[7];  const int* prev_d = (const int*)d_in[8];
  const int* succ_s = (const int*)d_in[9];  const int* succ_d = (const int*)d_in[10];
  const int* link_s = (const int*)d_in[11]; const int* link_d = (const int*)d_in[12];
  const int* plc_s  = (const int*)d_in[13]; const int* plc_d  = (const int*)d_in[14];
  const int* srv_s  = (const int*)d_in[15]; const int* srv_d  = (const int*)d_in[16];
  const float* params = (const float*)d_in[17];
  float* out = (float*)d_out;

  char* base = (char*)d_ws; size_t off = 0;
  auto alloc = [&](size_t bytes) -> void* {
    off = (off + 255) & ~(size_t)255;
    void* p = base + off; off += bytes; return p;
  };
  float* opf_a   = (float*)alloc((size_t)N_OPS * 40 * 4);
  float* opf_b   = (float*)alloc((size_t)N_OPS * 40 * 4);
  float* acc_op  = (float*)alloc((size_t)N_OPS * 40 * 4);
  float* devf_a  = (float*)alloc((size_t)N_DEV * 40 * 4);
  float* devf_b  = (float*)alloc((size_t)N_DEV * 40 * 4);
  float* acc_dev = (float*)alloc((size_t)N_DEV * 40 * 4);
  unsigned short* opbf_a  = (unsigned short*)alloc((size_t)N_OPS * 40 * 2);
  unsigned short* opbf_b  = (unsigned short*)alloc((size_t)N_OPS * 40 * 2);
  unsigned short* devbf_a = (unsigned short*)alloc((size_t)N_DEV * 40 * 2);
  unsigned short* devbf_b = (unsigned short*)alloc((size_t)N_DEV * 40 * 2);
  unsigned short* ef_link  = (unsigned short*)alloc((size_t)E_LINK  * 8 * 2);
  unsigned short* ef_prev  = (unsigned short*)alloc((size_t)E_PREV  * 8 * 2);
  unsigned short* ef_succ  = (unsigned short*)alloc((size_t)E_SUCC  * 8 * 2);
  unsigned short* ef_place = (unsigned short*)alloc((size_t)E_PLACE * 8 * 2);
  unsigned short* ef_serve = (unsigned short*)alloc((size_t)E_SERVE * 8 * 2);
  float* inv_prev  = (float*)alloc((size_t)N_OPS * 4);
  float* inv_succ  = (float*)alloc((size_t)N_OPS * 4);
  float* inv_serve = (float*)alloc((size_t)N_OPS * 4);
  float* inv_link  = (float*)alloc((size_t)N_DEV * 4);
  float* inv_place = (float*)alloc((size_t)N_DEV * 4);
  unsigned short* wpack = (unsigned short*)alloc((size_t)120 * 512 * 2);

  // per-etype in-degree -> reciprocal (counts reuse inv buffers as u32)
  hipMemsetAsync(inv_prev,  0, (size_t)N_OPS * 4, stream);
  hipMemsetAsync(inv_succ,  0, (size_t)N_OPS * 4, stream);
  hipMemsetAsync(inv_serve, 0, (size_t)N_OPS * 4, stream);
  hipMemsetAsync(inv_link,  0, (size_t)N_DEV * 4, stream);
  hipMemsetAsync(inv_place, 0, (size_t)N_DEV * 4, stream);
  count_kernel<<<(E_PREV  + 255) / 256, 256, 0, stream>>>(prev_d, (unsigned int*)inv_prev,  E_PREV);
  count_kernel<<<(E_SUCC  + 255) / 256, 256, 0, stream>>>(succ_d, (unsigned int*)inv_succ,  E_SUCC);
  count_kernel<<<(E_SERVE + 255) / 256, 256, 0, stream>>>(srv_d,  (unsigned int*)inv_serve, E_SERVE);
  count_kernel<<<(E_LINK  + 255) / 256, 256, 0, stream>>>(link_d, (unsigned int*)inv_link,  E_LINK);
  count_kernel<<<(E_PLACE + 255) / 256, 256, 0, stream>>>(plc_d,  (unsigned int*)inv_place, E_PLACE);
  invert_kernel<<<(N_OPS + 255) / 256, 256, 0, stream>>>(inv_prev,  N_OPS);
  invert_kernel<<<(N_OPS + 255) / 256, 256, 0, stream>>>(inv_succ,  N_OPS);
  invert_kernel<<<(N_OPS + 255) / 256, 256, 0, stream>>>(inv_serve, N_OPS);
  invert_kernel<<<1, 256, 0, stream>>>(inv_link,  N_DEV);
  invert_kernel<<<1, 256, 0, stream>>>(inv_place, N_DEV);

  pack_weights_kernel<<<120, 32, 0, stream>>>(params, wpack);

  edge_trans_kernel<<<(E_LINK  + 255) / 256, 256, 0, stream>>>(link_f,   params, P_ET_LINK_W,  P_ET_LINK_B,  12, ef_link,  E_LINK);
  edge_trans_kernel<<<(E_PREV  + 255) / 256, 256, 0, stream>>>(tensor_f, params, P_ET_PREV_W,  P_ET_PREV_B,   8, ef_prev,  E_PREV);
  edge_trans_kernel<<<(E_SUCC  + 255) / 256, 256, 0, stream>>>(tensor_f, params, P_ET_SUCC_W,  P_ET_SUCC_B,   8, ef_succ,  E_SUCC);
  edge_trans_kernel<<<(E_PLACE + 255) / 256, 256, 0, stream>>>(place_f,  params, P_ET_PLACE_W, P_ET_PLACE_B,  4, ef_place, E_PLACE);
  edge_trans_kernel<<<(E_SERVE + 255) / 256, 256, 0, stream>>>(place_f,  params, P_ET_SERVE_W, P_ET_SERVE_B,  4, ef_serve, E_SERVE);

  op_trans_kernel<<<(N_OPS + 255) / 256, 256, 0, stream>>>(op_feats, ntp, op_types, params, opf_a, opbf_a, N_OPS);
  dev_trans_kernel<<<1, 256, 0, stream>>>(dev_feats, params, devf_a, devbf_a);

  float* opc = opf_a; float* opn = opf_b;
  float* dvc = devf_a; float* dvn = devf_b;
  unsigned short* opbc = opbf_a; unsigned short* opbn = opbf_b;
  unsigned short* dvbc = devbf_a; unsigned short* dvbn = devbf_b;
  const int posmap[5] = {0, 2, 4, 1, 3}; // et(link,prev,succ,place,serve) -> sorted param position

  for (int l = 0; l < 4; ++l) {
    hipMemsetAsync(acc_op,  0, (size_t)N_OPS * 40 * 4, stream);
    hipMemsetAsync(acc_dev, 0, (size_t)N_DEV * 40 * 4, stream);
    struct Cfg { const unsigned short* src; const unsigned short* ef; const int* s; const int* d;
                 const float* inv; float* acc; int E; };
    const Cfg cfg[5] = {
      { dvbc, ef_link,  link_s, link_d, inv_link,  acc_dev, E_LINK  },
      { opbc, ef_prev,  prev_s, prev_d, inv_prev,  acc_op,  E_PREV  },
      { opbc, ef_succ,  succ_s, succ_d, inv_succ,  acc_op,  E_SUCC  },
      { opbc, ef_place, plc_s,  plc_d,  inv_place, acc_dev, E_PLACE },
      { dvbc, ef_serve, srv_s,  srv_d,  inv_serve, acc_op,  E_SERVE },
    };
    for (int et = 0; et < 5; ++et) {
      const float* bias = params + P_GCONV + l * 9800 + posmap[et] * 1960;
      const unsigned short* wp = wpack + (size_t)((l * 5 + et) * 6) * 512;
      int blocks = (cfg[et].E + 127) / 128;   // 8 waves/block, 16 edges/wave
      gconv_gemm_kernel<<<blocks, 256, 0, stream>>>(cfg[et].src, cfg[et].ef, cfg[et].s, cfg[et].d,
                                                    cfg[et].inv, wp, bias, cfg[et].acc, cfg[et].E);
    }
    int do_elu = (l < 3) ? 1 : 0;
    combine_kernel<<<(N_OPS * 40 + 255) / 256, 256, 0, stream>>>(opc, acc_op, opn, opbn, 1.f / 3.f, N_OPS * 40, do_elu);
    combine_kernel<<<(N_DEV * 40 + 255) / 256, 256, 0, stream>>>(dvc, acc_dev, dvn, dvbn, 0.5f, N_DEV * 40, do_elu);
    float* t = opc; opc = opn; opn = t;
    t = dvc; dvc = dvn; dvn = t;
    unsigned short* tb = opbc; opbc = opbn; opbn = tb;
    tb = dvbc; dvbc = dvbn; dvbn = tb;
  }

  final_kernel<<<(E_PLACE + 255) / 256, 256, 0, stream>>>(opc, dvc, place_f, plc_s, plc_d, params, out, E_PLACE);
}
